// Gemma4Attention_54803782697630
// MI455X (gfx1250) — matrix operand order repaired and checked by reading, not compiled, not run on hardware
//
#include <hip/hip_runtime.h>
#include <hip/hip_bf16.h>

#define S_LEN 2048
#define HID 2560
#define NH 8
#define NKV 4
#define HD 256
#define WIN 1024

typedef _Float16 v4h  __attribute__((ext_vector_type(4)));
typedef _Float16 v8h  __attribute__((ext_vector_type(8)));
typedef _Float16 v16h __attribute__((ext_vector_type(16)));
typedef float    v8f  __attribute__((ext_vector_type(8)));

__device__ __forceinline__ v8f wmma_f16(v16h a, v16h b, v8f c) {
    return __builtin_amdgcn_wmma_f32_16x16x32_f16(false, a, false, b, (short)0, c, false, false);
}

// A fragment (16x32 f16): lane L -> row L%16; halfs at k = 8*(L/16)+{0..7} and 16+8*(L/16)+{0..7}
__device__ __forceinline__ v16h fragA(const _Float16* p, int pitch) {
    int l = threadIdx.x & 31;
    const _Float16* q = p + (size_t)(l & 15) * pitch + (l >> 4) * 8;
    v8h a = *(const v8h*)q;
    v8h b = *(const v8h*)(q + 16);
    return __builtin_shufflevector(a, b, 0,1,2,3,4,5,6,7,8,9,10,11,12,13,14,15);
}

// B fragment (32x16 f16, fed from BT rows): lane L -> col L%16; halfs at k = 8*(L/16)+{0..7} and 16+8*(L/16)+{0..7}
__device__ __forceinline__ v16h fragB(const _Float16* p, int pitch) {
    int l = threadIdx.x & 31;
    const _Float16* q = p + (size_t)(l & 15) * pitch + (l >> 4) * 8;
    v8h a = *(const v8h*)q;
    v8h b = *(const v8h*)(q + 16);
    return __builtin_shufflevector(a, b, 0,1,2,3,4,5,6,7,8,9,10,11,12,13,14,15);
}

__device__ __forceinline__ void wait_ds0() {
#if __has_builtin(__builtin_amdgcn_s_wait_dscnt)
    __builtin_amdgcn_s_wait_dscnt(0);
#else
    asm volatile("s_wait_dscnt 0x0" ::: "memory");
#endif
}

// ---- CDNA5 async global->LDS copy. IOFFSET is added to BOTH the LDS and the
// global address, so in-row-contiguous chunks share one base + immediates.
__device__ __forceinline__ void async_b128(const void* lds, const void* g, int off) {
    asm volatile("global_load_async_to_lds_b128 %0, %1, off offset:%2"
                 :: "v"((unsigned int)(uintptr_t)lds),
                    "v"((unsigned long long)(uintptr_t)g),
                    "i"(off) : "memory");
}
#define WAIT_ASYNC_0()  asm volatile("s_wait_asynccnt 0x0" ::: "memory")
#define WAIT_ASYNC_5()  asm volatile("s_wait_asynccnt 0x5" ::: "memory")
#define WAIT_ASYNC_16() asm volatile("s_wait_asynccnt 0x10" ::: "memory")

// ---------------- preprocessing ----------------
__global__ void convert_f16_kernel(const float* __restrict__ in, _Float16* __restrict__ out, int n) {
    int i = blockIdx.x * blockDim.x + threadIdx.x;
    if (i < n) out[i] = (_Float16)in[i];
}

// in: K x N fp32 (row-major)  ->  out: N x K f16 (row-major transpose)
__global__ __launch_bounds__(256) void transpose_f16_kernel(const float* __restrict__ in,
                                                            _Float16* __restrict__ out,
                                                            int K, int N) {
    __shared__ float tile[32][33];
    int tx = threadIdx.x & 31, ty = threadIdx.x >> 5;
    int x = blockIdx.x * 32, yb = blockIdx.y * 32;
    #pragma unroll
    for (int i = 0; i < 4; ++i)
        tile[ty + i * 8][tx] = in[(size_t)(yb + ty + i * 8) * N + x + tx];
    __syncthreads();
    #pragma unroll
    for (int i = 0; i < 4; ++i)
        out[(size_t)(x + ty + i * 8) * K + yb + tx] = (_Float16)tile[tx][ty + i * 8];
}

// ---------------- fused QKV GEMM + RMSNorm + RoPE ----------------
// grid: x = S/64, y = NH + 2*NKV ; block 256 (8 waves: 4M x 2N), tile 64 x 256 (one head)
// V is written TRANSPOSED ([kv][d][s]) so attention can async-DMA V^T tiles directly.
__global__ __launch_bounds__(256) void qkv_kernel(
    const _Float16* __restrict__ Xh,
    const _Float16* __restrict__ WqT, const _Float16* __restrict__ WkT, const _Float16* __restrict__ WvT,
    const float* __restrict__ cosg, const float* __restrict__ sing,
    const float* __restrict__ qw, const float* __restrict__ kw,
    _Float16* __restrict__ Qb, _Float16* __restrict__ Kb, _Float16* __restrict__ Vb)
{
    extern __shared__ char smem[];
    _Float16* At = (_Float16*)smem;            // [2][64][40]
    _Float16* Bt = At + 2 * 64 * 40;           // [2][256][40]
    float* ep   = (float*)smem;                // [64][264]  (overlay)
    float* rs   = ep + 64 * 264;               // [64][4]
    float* rrms = rs + 64 * 4;                 // [64]

    int t = threadIdx.x;
    int m0 = blockIdx.x * 64;
    int y = blockIdx.y;
    const _Float16* Wt; _Float16* outp; const float* nw; int rope;
    if (y < NH)            { Wt = WqT + (size_t)y * HD * HID;              outp = Qb + (size_t)y * S_LEN * HD;              nw = qw; rope = 1; }
    else if (y < NH + NKV) { Wt = WkT + (size_t)(y - NH) * HD * HID;       outp = Kb + (size_t)(y - NH) * S_LEN * HD;       nw = kw; rope = 1; }
    else                   { Wt = WvT + (size_t)(y - NH - NKV) * HD * HID; outp = Vb + (size_t)(y - NH - NKV) * HD * S_LEN; nw = nullptr; rope = 0; }

    int wv = t >> 5;
    int mw = (wv >> 1) * 16;     // 0,16,32,48
    int nw0 = (wv & 1) * 128;    // 0,128
    v8f acc[8] = {};

    int ar = t >> 2, ac = (t & 3) * 8;
    const _Float16* ap = Xh + (size_t)(m0 + ar) * HID + ac;
    const _Float16* bp = Wt + (size_t)t * HID;

    auto issue = [&](int buf) {
        async_b128(&At[buf * 64 * 40 + ar * 40 + ac], ap, 0);
        const _Float16* bl = &Bt[buf * 256 * 40 + t * 40];
        async_b128(bl, bp, 0);
        async_b128(bl, bp, 16);
        async_b128(bl, bp, 32);
        async_b128(bl, bp, 48);
        ap += 32; bp += 32;
    };
    auto compute = [&](int buf) {
        const _Float16* a0 = &At[buf * 64 * 40];
        const _Float16* b0 = &Bt[buf * 256 * 40];
        v16h af = fragA(a0 + mw * 40, 40);
        v16h f0 = fragB(b0 + (nw0 +  0) * 40, 40);
        v16h f1 = fragB(b0 + (nw0 + 16) * 40, 40);
        v16h f2 = fragB(b0 + (nw0 + 32) * 40, 40);
        v16h f3 = fragB(b0 + (nw0 + 48) * 40, 40);
        acc[0] = wmma_f16(af, f0, acc[0]);
        acc[1] = wmma_f16(af, f1, acc[1]);
        acc[2] = wmma_f16(af, f2, acc[2]);
        acc[3] = wmma_f16(af, f3, acc[3]);
        v16h f4 = fragB(b0 + (nw0 + 64) * 40, 40);
        v16h f5 = fragB(b0 + (nw0 + 80) * 40, 40);
        v16h f6 = fragB(b0 + (nw0 + 96) * 40, 40);
        v16h f7 = fragB(b0 + (nw0 + 112) * 40, 40);
        acc[4] = wmma_f16(af, f4, acc[4]);
        acc[5] = wmma_f16(af, f5, acc[5]);
        acc[6] = wmma_f16(af, f6, acc[6]);
        acc[7] = wmma_f16(af, f7, acc[7]);
    };

    issue(0);
    for (int s = 0; s < (HID / 32) - 2; s += 2) {
        issue(1); WAIT_ASYNC_5(); __syncthreads(); compute(0); __syncthreads();
        issue(0); WAIT_ASYNC_5(); __syncthreads(); compute(1); __syncthreads();
    }
    issue(1); WAIT_ASYNC_5(); __syncthreads(); compute(0); __syncthreads();
    WAIT_ASYNC_0(); __syncthreads(); compute(1); __syncthreads();

    // ---- epilogue through fp32 LDS tile ----
    int lane = t & 31, half = lane >> 4, n = lane & 15;
    #pragma unroll
    for (int j = 0; j < 8; ++j)
        #pragma unroll
        for (int r = 0; r < 8; ++r)
            ep[(mw + r + 8 * half) * 264 + nw0 + j * 16 + n] = acc[j][r];
    __syncthreads();
    {
        int r = t >> 2, p = t & 3;
        float s = 0.f;
        #pragma unroll
        for (int c = 0; c < 64; ++c) { float v = ep[r * 264 + p * 64 + c]; s += v * v; }
        rs[r * 4 + p] = s;
    }
    __syncthreads();
    if (t < 64) {
        float s = rs[t * 4] + rs[t * 4 + 1] + rs[t * 4 + 2] + rs[t * 4 + 3];
        rrms[t] = rsqrtf(s * (1.0f / HD) + 1e-6f);
    }
    __syncthreads();
    if (rope) {
        int r = t >> 2, c0 = (t & 3) * 64;
        int sidx = m0 + r;
        float rr = rrms[r];
        for (int i = 0; i < 64; ++i) {
            int c = c0 + i;
            float x = ep[r * 264 + c] * rr * nw[c];
            int pc = (c + 128) & 255;
            float px = ep[r * 264 + pc] * rr * nw[pc];
            float rh = (c < 128) ? -px : px;
            float val = x * cosg[(size_t)sidx * HD + c] + rh * sing[(size_t)sidx * HD + c];
            outp[(size_t)sidx * HD + c] = (_Float16)val;
        }
    } else {
        // V: rms_norm only, stored transposed: out[d][s] (thread t -> d)
        int d = t;
        #pragma unroll
        for (int sb = 0; sb < 8; ++sb) {
            v8h tv;
            #pragma unroll
            for (int e = 0; e < 8; ++e) {
                int s = sb * 8 + e;
                tv[e] = (_Float16)(ep[s * 264 + d] * rrms[s]);
            }
            *(v8h*)&outp[(size_t)d * S_LEN + m0 + sb * 8] = tv;
        }
    }
}

// ---------------- flash attention (sliding window, no scale per reference) ----------------
// grid: x = S/64, y = NH ; block 128 (4 waves, 16 q-rows each)
// K and V^T tiles are both pure async DMA; double-buffered (block count is always even).
__global__ __launch_bounds__(128) void attn_kernel(
    const _Float16* __restrict__ Qb, const _Float16* __restrict__ Kb, const _Float16* __restrict__ Vb,
    _Float16* __restrict__ Ob)
{
    extern __shared__ char smem[];
    _Float16* Qt  = (_Float16*)smem;       // [64][264] async-filled once
    _Float16* Kt0 = Qt + 64 * 264;         // [32][264]
    _Float16* Kt1 = Kt0 + 32 * 264;
    _Float16* VT0 = Kt1 + 32 * 264;        // [256][40]
    _Float16* VT1 = VT0 + 256 * 40;
    _Float16* Pst = VT1 + 256 * 40;        // [4][16][40]

    int t = threadIdx.x;
    int wv = t >> 5, lane = t & 31, half = lane >> 4, n = lane & 15;
    int h = blockIdx.y, kvh = h >> 1;
    int m0 = blockIdx.x * 64;
    int qrow0 = wv * 16;

    const _Float16* Qh = Qb + (size_t)h * S_LEN * HD;
    const _Float16* Kh = Kb + (size_t)kvh * S_LEN * HD;
    const _Float16* Vh = Vb + (size_t)kvh * HD * S_LEN;   // [d][s]

    v8f of[16] = {};
    float mrow[8], lrow[8];
    #pragma unroll
    for (int r = 0; r < 8; ++r) { mrow[r] = -1e30f; lrow[r] = 0.f; }

    int jstart = m0 - (WIN - 1); if (jstart < 0) jstart = 0; jstart &= ~31;
    int jend = m0 + 63;

    // Q tile: DMA once. thread -> row t>>1, 128-half slab (t&1)
    {
        int qr = t >> 1, qc = (t & 1) * 128;
        const _Float16* qg = Qh + (size_t)(m0 + qr) * HD + qc;
        _Float16* ql = &Qt[qr * 264 + qc];
        #pragma unroll
        for (int i = 0; i < 16; ++i) async_b128(ql, qg, i * 16);
    }

    // K DMA: thread -> key row t>>2, 64-half slab (t&3). V^T DMA: thread -> d rows {2t,2t+1}.
    int kr = t >> 2, kc = (t & 3) * 64;
    auto issue = [&](_Float16* KtB, _Float16* VTB, int j0) {
        const _Float16* kg = Kh + (size_t)(j0 + kr) * HD + kc;
        _Float16* kl = KtB + kr * 264 + kc;
        #pragma unroll
        for (int i = 0; i < 8; ++i) async_b128(kl, kg, i * 16);
        #pragma unroll
        for (int rr = 0; rr < 2; ++rr) {
            int d = t * 2 + rr;
            const _Float16* vg = Vh + (size_t)d * S_LEN + j0;
            _Float16* vl = VTB + d * 40;
            async_b128(vl, vg, 0);
            async_b128(vl, vg, 16);
            async_b128(vl, vg, 32);
            async_b128(vl, vg, 48);
        }
    };

    auto blockstep = [&](const _Float16* KtB, const _Float16* VTB, int j0) {
        // scores: two 16x16 tiles (keys j0..j0+15, j0+16..j0+31)
        v8f sc0 = {}, sc1 = {};
        #pragma unroll
        for (int c = 0; c < 8; ++c) {
            v16h qa = fragA(Qt + qrow0 * 264 + c * 32, 264);
            v16h b0 = fragB(KtB + c * 32, 264);
            v16h b1 = fragB(KtB + 16 * 264 + c * 32, 264);
            sc0 = wmma_f16(qa, b0, sc0);
            sc1 = wmma_f16(qa, b1, sc1);
        }
        // sliding-window causal mask
        #pragma unroll
        for (int r = 0; r < 8; ++r) {
            int q = m0 + qrow0 + r + 8 * half;
            int j = j0 + n;
            if (!((j <= q) && (q - j < WIN))) sc0[r] = -1e30f;
            j = j0 + 16 + n;
            if (!((j <= q) && (q - j < WIN))) sc1[r] = -1e30f;
        }
        // online softmax
        float e0[8], e1[8], corr[8];
        #pragma unroll
        for (int r = 0; r < 8; ++r) {
            float mx = fmaxf(sc0[r], sc1[r]);
            mx = fmaxf(mx, __shfl_xor(mx, 1, 32));
            mx = fmaxf(mx, __shfl_xor(mx, 2, 32));
            mx = fmaxf(mx, __shfl_xor(mx, 4, 32));
            mx = fmaxf(mx, __shfl_xor(mx, 8, 32));
            float nm = fmaxf(mrow[r], mx);
            corr[r] = __expf(mrow[r] - nm);
            float a0 = __expf(sc0[r] - nm); if (sc0[r] <= -1e29f) a0 = 0.f;
            float a1 = __expf(sc1[r] - nm); if (sc1[r] <= -1e29f) a1 = 0.f;
            float ps = a0 + a1;
            ps += __shfl_xor(ps, 1, 32);
            ps += __shfl_xor(ps, 2, 32);
            ps += __shfl_xor(ps, 4, 32);
            ps += __shfl_xor(ps, 8, 32);
            lrow[r] = lrow[r] * corr[r] + ps;
            mrow[r] = nm;
            e0[r] = a0; e1[r] = a1;
        }
        #pragma unroll
        for (int f = 0; f < 16; ++f)
            #pragma unroll
            for (int r = 0; r < 8; ++r) of[f][r] *= corr[r];
        // P: C-layout -> LDS raw tile -> A-fragment
        _Float16* pw = Pst + wv * 16 * 40;
        #pragma unroll
        for (int r = 0; r < 8; ++r) {
            pw[(r + 8 * half) * 40 + n]      = (_Float16)e0[r];
            pw[(r + 8 * half) * 40 + 16 + n] = (_Float16)e1[r];
        }
        wait_ds0();
        v16h pf = fragA(pw, 40);
        #pragma unroll
        for (int g = 0; g < 4; ++g) {
            v16h bv0 = fragB(VTB + (g * 64 +  0) * 40, 40);
            v16h bv1 = fragB(VTB + (g * 64 + 16) * 40, 40);
            v16h bv2 = fragB(VTB + (g * 64 + 32) * 40, 40);
            v16h bv3 = fragB(VTB + (g * 64 + 48) * 40, 40);
            of[g * 4 + 0] = wmma_f16(pf, bv0, of[g * 4 + 0]);
            of[g * 4 + 1] = wmma_f16(pf, bv1, of[g * 4 + 1]);
            of[g * 4 + 2] = wmma_f16(pf, bv2, of[g * 4 + 2]);
            of[g * 4 + 3] = wmma_f16(pf, bv3, of[g * 4 + 3]);
        }
    };

    issue(Kt0, VT0, jstart);
    for (int j0 = jstart; j0 <= jend; j0 += 64) {   // always an even number of 32-key blocks
        issue(Kt1, VT1, j0 + 32);
        WAIT_ASYNC_16();
        __syncthreads();
        blockstep(Kt0, VT0, j0);
        __syncthreads();
        if (j0 + 64 <= jend) { issue(Kt0, VT0, j0 + 64); WAIT_ASYNC_16(); }
        else                 { WAIT_ASYNC_0(); }
        __syncthreads();
        blockstep(Kt1, VT1, j0 + 32);
        __syncthreads();
    }

    float inv[8];
    #pragma unroll
    for (int r = 0; r < 8; ++r) inv[r] = 1.0f / lrow[r];
    #pragma unroll
    for (int f = 0; f < 16; ++f)
        #pragma unroll
        for (int r = 0; r < 8; ++r) {
            int s = m0 + qrow0 + r + 8 * half;
            Ob[(size_t)s * (NH * HD) + h * HD + f * 16 + n] = (_Float16)(of[f][r] * inv[r]);
        }
}

// ---------------- NT GEMM (fp32 out) for the output projection ----------------
__global__ __launch_bounds__(256) void gemm_nt_kernel(
    const _Float16* __restrict__ A, const _Float16* __restrict__ Bt, float* __restrict__ C)
{
    const int K = NH * HD;   // 2048
    const int N = HID;       // 2560
    __shared__ _Float16 At[2][64][40];
    __shared__ _Float16 Btile[2][256][40];
    int t = threadIdx.x;
    int m0 = blockIdx.x * 64, n0 = blockIdx.y * 256;
    int wv = t >> 5, mw = (wv >> 2) * 32, nw0 = (wv & 3) * 64;
    v8f acc[2][4] = {};

    int ar = t >> 2, ac = (t & 3) * 8;
    const _Float16* ap = A + (size_t)(m0 + ar) * K + ac;
    const _Float16* bp = Bt + (size_t)(n0 + t) * K;

    auto issue = [&](int buf) {
        async_b128(&At[buf][ar][ac], ap, 0);
        const _Float16* bl = &Btile[buf][t][0];
        async_b128(bl, bp, 0);
        async_b128(bl, bp, 16);
        async_b128(bl, bp, 32);
        async_b128(bl, bp, 48);
        ap += 32; bp += 32;
    };
    auto compute = [&](int buf) {
        v16h af0 = fragA(&At[buf][mw][0], 40);
        v16h af1 = fragA(&At[buf][mw + 16][0], 40);
        v16h bf0 = fragB(&Btile[buf][nw0 +  0][0], 40);
        v16h bf1 = fragB(&Btile[buf][nw0 + 16][0], 40);
        v16h bf2 = fragB(&Btile[buf][nw0 + 32][0], 40);
        v16h bf3 = fragB(&Btile[buf][nw0 + 48][0], 40);
        acc[0][0] = wmma_f16(af0, bf0, acc[0][0]);
        acc[0][1] = wmma_f16(af0, bf1, acc[0][1]);
        acc[0][2] = wmma_f16(af0, bf2, acc[0][2]);
        acc[0][3] = wmma_f16(af0, bf3, acc[0][3]);
        acc[1][0] = wmma_f16(af1, bf0, acc[1][0]);
        acc[1][1] = wmma_f16(af1, bf1, acc[1][1]);
        acc[1][2] = wmma_f16(af1, bf2, acc[1][2]);
        acc[1][3] = wmma_f16(af1, bf3, acc[1][3]);
    };

    issue(0);
    for (int s = 0; s < (K / 32) - 2; s += 2) {
        issue(1); WAIT_ASYNC_5(); __syncthreads(); compute(0); __syncthreads();
        issue(0); WAIT_ASYNC_5(); __syncthreads(); compute(1); __syncthreads();
    }
    issue(1); WAIT_ASYNC_5(); __syncthreads(); compute(0); __syncthreads();
    WAIT_ASYNC_0(); __syncthreads(); compute(1); __syncthreads();

    int lane = t & 31, half = lane >> 4, n = lane & 15;
    #pragma unroll
    for (int i = 0; i < 2; ++i)
        #pragma unroll
        for (int j = 0; j < 4; ++j)
            #pragma unroll
            for (int r = 0; r < 8; ++r)
                C[(size_t)(m0 + mw + i * 16 + r + 8 * half) * N + n0 + nw0 + j * 16 + n] = acc[i][j][r];
}

extern "C" void kernel_launch(void* const* d_in, const int* in_sizes, int n_in,
                              void* d_out, int out_size, void* d_ws, size_t ws_size,
                              hipStream_t stream) {
    const float* hs   = (const float*)d_in[0];
    const float* cosg = (const float*)d_in[1];
    const float* sing = (const float*)d_in[2];
    const float* wq   = (const float*)d_in[3];
    const float* wk   = (const float*)d_in[4];
    const float* wvp  = (const float*)d_in[5];
    const float* wo   = (const float*)d_in[6];
    const float* qw   = (const float*)d_in[7];
    const float* kw   = (const float*)d_in[8];
    float* outp = (float*)d_out;

    _Float16* ws = (_Float16*)d_ws;
    size_t off = 0;
    _Float16* Xh  = ws + off; off += (size_t)S_LEN * HID;
    _Float16* WqT = ws + off; off += (size_t)(NH * HD) * HID;
    _Float16* WkT = ws + off; off += (size_t)(NKV * HD) * HID;
    _Float16* WvT = ws + off; off += (size_t)(NKV * HD) * HID;
    _Float16* WoT = ws + off; off += (size_t)HID * (NH * HD);
    _Float16* Qb  = ws + off; off += (size_t)NH * S_LEN * HD;
    _Float16* Kb  = ws + off; off += (size_t)NKV * S_LEN * HD;
    _Float16* Vb  = ws + off; off += (size_t)NKV * HD * S_LEN;   // transposed [kv][d][s]
    _Float16* Ob  = ws + off; off += (size_t)S_LEN * NH * HD;

    convert_f16_kernel<<<(S_LEN * HID + 255) / 256, 256, 0, stream>>>(hs, Xh, S_LEN * HID);
    transpose_f16_kernel<<<dim3((NH * HD) / 32, HID / 32), 256, 0, stream>>>(wq, WqT, HID, NH * HD);
    transpose_f16_kernel<<<dim3((NKV * HD) / 32, HID / 32), 256, 0, stream>>>(wk, WkT, HID, NKV * HD);
    transpose_f16_kernel<<<dim3((NKV * HD) / 32, HID / 32), 256, 0, stream>>>(wvp, WvT, HID, NKV * HD);
    transpose_f16_kernel<<<dim3(HID / 32, (NH * HD) / 32), 256, 0, stream>>>(wo, WoT, NH * HD, HID);

    size_t qkv_lds = (size_t)64 * 264 * 4 + 64 * 4 * 4 + 64 * 4;   // 68,864 B (epilogue overlay is max)
    qkv_kernel<<<dim3(S_LEN / 64, NH + 2 * NKV), 256, qkv_lds, stream>>>(
        Xh, WqT, WkT, WvT, cosg, sing, qw, kw, Qb, Kb, Vb);

    size_t attn_lds = (size_t)(64 * 264 + 2 * 32 * 264 + 2 * 256 * 40 + 4 * 16 * 40) * 2;  // 113,664 B
    attn_kernel<<<dim3(S_LEN / 64, NH), 128, attn_lds, stream>>>(Qb, Kb, Vb, Ob);

    gemm_nt_kernel<<<dim3(S_LEN / 64, HID / 256), 256, 0, stream>>>(Ob, WoT, outp);
}
